// QKVAttention_14456859918582
// MI455X (gfx1250) — matrix operand order repaired and checked by reading, not compiled, not run on hardware
//
#include <hip/hip_runtime.h>

#define SLN   1024
#define DIM   64
#define NH    16
#define KB    32      // keys per streaming step
#define NBAND 48      // band rows: qi-ki+31 for qi in [0,16), ki in [0,32)

typedef __attribute__((ext_vector_type(16))) __bf16 v16bf;
typedef __attribute__((ext_vector_type(8)))  float  v8f;

union AFrag {
    v16bf v;
    uint4 q2[2];
    unsigned short s[16];
};

// Per-wave private LDS (no cross-wave sharing -> no barriers needed).
struct WaveLds {
    unsigned short Qb[16 * 64];      // Q tile bf16, row-major [q][d]           2048 B
    unsigned short Kb[KB * 64];      // K tile bf16, row-major [key][d]         4096 B
    unsigned short Vt[64 * KB];      // V^T  bf16 [d][key]                      4096 B
    unsigned short Ak[NBAND * 64];   // ak band rows [j][d]                     6144 B
    unsigned short AvT[64 * 64];     // av band transposed, padded [d][j:64]    8192 B
    float          Ms[16 * NBAND];   // M = Q @ Tband^T staging                 3072 B
    unsigned short Pb[16 * KB];      // probs bf16 [q][k]                       1024 B
};                                   // total 28672 B per wave, 2 waves = 56 KB

// Hardware packed f32->bf16 converter (1 VALU op per pair instead of ~7 in SW).
__device__ __forceinline__ unsigned int pk2(float a, float b) {
    unsigned int r;
    asm("v_cvt_pk_bf16_f32 %0, %1, %2" : "=v"(r) : "v"(a), "v"(b));
    return r;
}
__device__ __forceinline__ unsigned short f2bf(float a) {
    unsigned int r;
    asm("v_cvt_pk_bf16_f32 %0, %1, %1" : "=v"(r) : "v"(a));
    return (unsigned short)r;
}

__global__ __launch_bounds__(64, 1)
void relpos_flash_attn(const float* __restrict__ Q, const float* __restrict__ K,
                       const float* __restrict__ V, const float* __restrict__ mask,
                       const float* __restrict__ ak, const float* __restrict__ av,
                       float* __restrict__ out)
{
    __shared__ __align__(64) WaveLds lds[2];
    const int lane   = threadIdx.x & 31;
    const int wv     = threadIdx.x >> 5;
    WaveLds& w = lds[wv];

    const int tile = blockIdx.x;                  // BSP * (SL/32) blocks
    const int bh   = tile >> 5;                   // / (SL/32)
    const int q0   = ((tile & 31) << 5) + (wv << 4);
    const int b    = bh >> 4;                     // / NH
    const int lanelo = lane & 15;
    const int hi     = lane >> 4;
    const int hi8    = hi << 3;
    const int hi16   = hi << 4;

    // ---- stage Q tile as bf16 (each lane: half a row) ----
    {
        const int qi  = lane >> 1;
        const int dd0 = (lane & 1) << 5;
        const float4* src = (const float4*)(Q + ((long)bh * SLN + q0 + qi) * DIM + dd0);
        unsigned int* dst = (unsigned int*)&w.Qb[qi * 64 + dd0];
#pragma unroll
        for (int i = 0; i < 8; ++i) {
            float4 f = src[i];
            dst[2 * i]     = pk2(f.x, f.y);
            dst[2 * i + 1] = pk2(f.z, f.w);
        }
    }
    // ---- zero the j=48..63 pad of AvT once (A side is 0 there too) ----
    for (int i = lane; i < 64 * 16; i += 32) {
        int d = i >> 4, j = 48 + (i & 15);
        w.AvT[d * 64 + j] = 0;
    }
    asm volatile("s_wait_dscnt 0x0" ::: "memory");

    // ---- loop-invariant A fragments of Q (serve both QK^T and Q@Tband^T) ----
    AFrag aQ[2];
#pragma unroll
    for (int c = 0; c < 2; ++c) {
        int base = lanelo * 64 + c * 32 + hi16;
        aQ[c].q2[0] = *(const uint4*)&w.Qb[base];
        aQ[c].q2[1] = *(const uint4*)&w.Qb[base + 8];
    }

    v8f O[4];
    const v8f vzero = {0.f, 0.f, 0.f, 0.f, 0.f, 0.f, 0.f, 0.f};
#pragma unroll
    for (int t = 0; t < 4; ++t) O[t] = vzero;
    float mrun[8], lrun[8];
#pragma unroll
    for (int r = 0; r < 8; ++r) { mrun[r] = -3.0e38f; lrun[r] = 0.f; }

    const int rb0 = q0 + 1023 - 31;               // rbase = rb0 - k0
    const float scale = 0.125f;                   // D^-1/2

    for (int k0 = 0; k0 < SLN; k0 += KB) {
        // ---- stage K block, row-major bf16 [key][d] ----
        {
            const float4* src = (const float4*)(K + ((long)bh * SLN + k0 + lane) * DIM);
            unsigned int* dst = (unsigned int*)&w.Kb[lane * 64];
#pragma unroll
            for (int i = 0; i < 16; ++i) {
                float4 f = src[i];
                dst[2 * i]     = pk2(f.x, f.y);
                dst[2 * i + 1] = pk2(f.z, f.w);
            }
        }
        // ---- stage V block transposed, bf16 [d][key] ----
        {
            const float4* src = (const float4*)(V + ((long)bh * SLN + k0 + lane) * DIM);
#pragma unroll
            for (int i = 0; i < 16; ++i) {
                float4 f = src[i];
                int d = 4 * i;
                w.Vt[(d + 0) * KB + lane] = f2bf(f.x);
                w.Vt[(d + 1) * KB + lane] = f2bf(f.y);
                w.Vt[(d + 2) * KB + lane] = f2bf(f.z);
                w.Vt[(d + 3) * KB + lane] = f2bf(f.w);
            }
        }
        // ---- stage Toeplitz bands: T[r] = ak[max(r-1023,0), max(1023-r,0)] ----
        const int rbase = rb0 - k0;
        for (int j = lane; j < NBAND; j += 32) {
            int r  = rbase + j;
            bool ok = (r >= 0) && (r <= 2046);
            int rc = r < 0 ? 0 : (r > 2046 ? 2046 : r);
            int qr = (rc >= 1023) ? (rc - 1023) : 0;
            int kr = (rc >= 1023) ? 0 : (1023 - rc);
            long off = ((long)qr * SLN + kr) * DIM;
            const float4* sa = (const float4*)(ak + off);
            const float4* sv = (const float4*)(av + off);
            unsigned int* da = (unsigned int*)&w.Ak[j * 64];
#pragma unroll
            for (int i = 0; i < 16; ++i) {
                float4 fa = ok ? sa[i] : make_float4(0.f, 0.f, 0.f, 0.f);
                float4 fv = ok ? sv[i] : make_float4(0.f, 0.f, 0.f, 0.f);
                da[2 * i]     = pk2(fa.x, fa.y);
                da[2 * i + 1] = pk2(fa.z, fa.w);
                int d = 4 * i;
                w.AvT[(d + 0) * 64 + j] = f2bf(fv.x);
                w.AvT[(d + 1) * 64 + j] = f2bf(fv.y);
                w.AvT[(d + 2) * 64 + j] = f2bf(fv.z);
                w.AvT[(d + 3) * 64 + j] = f2bf(fv.w);
            }
        }
        // ---- prefetch next step's K/V rows and band rows (hide HBM latency) ----
        if (k0 + KB < SLN) {
            __builtin_prefetch(K + ((long)bh * SLN + k0 + KB + lane) * DIM, 0, 1);
            __builtin_prefetch(V + ((long)bh * SLN + k0 + KB + lane) * DIM, 0, 1);
            int rn = rbase - KB + lane;           // next band, rows j=lane
            int rnc = rn < 0 ? 0 : (rn > 2046 ? 2046 : rn);
            int qn = (rnc >= 1023) ? (rnc - 1023) : 0;
            int kn = (rnc >= 1023) ? 0 : (1023 - rnc);
            long offn = ((long)qn * SLN + kn) * DIM;
            __builtin_prefetch(ak + offn, 0, 1);
            __builtin_prefetch(av + offn, 0, 1);
        }
        asm volatile("s_wait_dscnt 0x0" ::: "memory");

        // ---- S = Q @ K^T (2 N-tiles) and M = Q @ Tband^T (3 N-tiles) ----
        v8f S0 = vzero, S1 = vzero, M0 = vzero, M1 = vzero, M2 = vzero;
#pragma unroll
        for (int c = 0; c < 2; ++c) {
            AFrag b0, b1, t0, t1, t2;
            b0.v = *(const v16bf*)&w.Kb[(lanelo)      * 64 + c * 32 + hi16];
            b1.v = *(const v16bf*)&w.Kb[(lanelo + 16) * 64 + c * 32 + hi16];
            t0.v = *(const v16bf*)&w.Ak[(lanelo)      * 64 + c * 32 + hi16];
            t1.v = *(const v16bf*)&w.Ak[(lanelo + 16) * 64 + c * 32 + hi16];
            t2.v = *(const v16bf*)&w.Ak[(lanelo + 32) * 64 + c * 32 + hi16];
            S0 = __builtin_amdgcn_wmma_f32_16x16x32_bf16(false, aQ[c].v, false, b0.v, (short)0, S0, false, false);
            S1 = __builtin_amdgcn_wmma_f32_16x16x32_bf16(false, aQ[c].v, false, b1.v, (short)0, S1, false, false);
            M0 = __builtin_amdgcn_wmma_f32_16x16x32_bf16(false, aQ[c].v, false, t0.v, (short)0, M0, false, false);
            M1 = __builtin_amdgcn_wmma_f32_16x16x32_bf16(false, aQ[c].v, false, t1.v, (short)0, M1, false, false);
            M2 = __builtin_amdgcn_wmma_f32_16x16x32_bf16(false, aQ[c].v, false, t2.v, (short)0, M2, false, false);
        }
        // stage M so the diagonal gather bias[qi,ki] = M[qi, qi-ki+31] is a trivial LDS read
#pragma unroll
        for (int r = 0; r < 8; ++r) {
            int row = r + hi8;
            w.Ms[row * NBAND + lanelo]      = M0[r];
            w.Ms[row * NBAND + lanelo + 16] = M1[r];
            w.Ms[row * NBAND + lanelo + 32] = M2[r];
        }
        asm volatile("s_wait_dscnt 0x0" ::: "memory");

        float mk0 = mask[b * SLN + k0 + lanelo];
        float mk1 = mask[b * SLN + k0 + 16 + lanelo];

        // ---- streaming softmax (rows live in 16-lane half-groups) ----
#pragma unroll
        for (int r = 0; r < 8; ++r) {
            int qi = r + hi8;
            float bias0 = w.Ms[qi * NBAND + (qi - lanelo + 31)];
            float bias1 = w.Ms[qi * NBAND + (qi - lanelo + 15)];
            float s0 = S0[r] * scale + bias0 - mk0;
            float s1 = S1[r] * scale + bias1 - mk1;
            float mx = fmaxf(s0, s1);
#pragma unroll
            for (int o = 1; o < 16; o <<= 1) mx = fmaxf(mx, __shfl_xor(mx, o, 32));
            float mn = fmaxf(mrun[r], mx);
            float al = __expf(mrun[r] - mn);
            mrun[r] = mn;
            float p0 = __expf(s0 - mn);
            float p1 = __expf(s1 - mn);
            float rs = p0 + p1;
#pragma unroll
            for (int o = 1; o < 16; o <<= 1) rs += __shfl_xor(rs, o, 32);
            lrun[r] = lrun[r] * al + rs;
            O[0][r] *= al; O[1][r] *= al; O[2][r] *= al; O[3][r] *= al;
            w.Pb[qi * KB + lanelo]      = f2bf(p0);
            w.Pb[qi * KB + lanelo + 16] = f2bf(p1);
        }
        asm volatile("s_wait_dscnt 0x0" ::: "memory");

        // ---- A fragments: probs, and band-scattered probs C[qi,j] = P[qi, qi+31-j] ----
        AFrag aP;
        {
            int base = lanelo * KB + hi16;
            aP.q2[0] = *(const uint4*)&w.Pb[base];
            aP.q2[1] = *(const uint4*)&w.Pb[base + 8];
        }
        AFrag aC[2];
#pragma unroll
        for (int c = 0; c < 2; ++c) {
#pragma unroll
            for (int e = 0; e < 16; ++e) {
                int j  = c * 32 + hi16 + e;
                int ki = lanelo + 31 - j;
                aC[c].s[e] = (ki >= 0 && ki < KB) ? w.Pb[lanelo * KB + ki]
                                                  : (unsigned short)0;
            }
        }
        // ---- O += P @ V  and  O += C @ Tv_band ----
#pragma unroll
        for (int t = 0; t < 4; ++t) {
            AFrag bV;
            bV.v = *(const v16bf*)&w.Vt[(lanelo + 16 * t) * KB + hi16];
            O[t] = __builtin_amdgcn_wmma_f32_16x16x32_bf16(false, aP.v, false, bV.v, (short)0, O[t], false, false);
#pragma unroll
            for (int c = 0; c < 2; ++c) {
                AFrag bA;
                bA.v = *(const v16bf*)&w.AvT[(lanelo + 16 * t) * 64 + c * 32 + hi16];
                O[t] = __builtin_amdgcn_wmma_f32_16x16x32_bf16(false, aC[c].v, false, bA.v, (short)0, O[t], false, false);
            }
        }
    }

    // ---- normalize and write out (fp32) ----
#pragma unroll
    for (int r = 0; r < 8; ++r) {
        int qi = r + hi8;
        float inv = 1.0f / lrun[r];
        float* dst = out + ((long)bh * SLN + q0 + qi) * DIM + lanelo;
        dst[0]  = O[0][r] * inv;
        dst[16] = O[1][r] * inv;
        dst[32] = O[2][r] * inv;
        dst[48] = O[3][r] * inv;
    }
}

extern "C" void kernel_launch(void* const* d_in, const int* in_sizes, int n_in,
                              void* d_out, int out_size, void* d_ws, size_t ws_size,
                              hipStream_t stream) {
    (void)in_sizes; (void)n_in; (void)out_size; (void)d_ws; (void)ws_size;
    const float* Q    = (const float*)d_in[0];
    const float* K    = (const float*)d_in[1];
    const float* V    = (const float*)d_in[2];
    const float* mask = (const float*)d_in[3];
    const float* ak   = (const float*)d_in[4];
    const float* av   = (const float*)d_in[5];
    float* out = (float*)d_out;

    dim3 grid(128 * (SLN / 32));   // BSP * q-tiles-of-32 (2 waves x 16 rows)
    dim3 block(64);                // 2 wave32
    relpos_flash_attn<<<grid, block, 0, stream>>>(Q, K, V, mask, ak, av, out);
}
